// ScaleDotProductAttention_57380763075016
// MI455X (gfx1250) — matrix operand order repaired and checked by reading, not compiled, not run on hardware
//
#include <hip/hip_runtime.h>

typedef __attribute__((ext_vector_type(16))) _Float16 v16h;
typedef __attribute__((ext_vector_type(8)))  _Float16 v8h;
typedef __attribute__((ext_vector_type(8)))  float    v8f;
typedef __attribute__((ext_vector_type(8)))  int      v8i;

#define S_LEN 2048
#define HEAD_D 64
#define KT 64          // keys per tile iteration
#define WAVES 8        // waves per block (256 threads)
#define LPAD 8         // LDS row padding (halves) to break bank conflicts
#define BH   32        // B*H

// ---- half-wave swap: lane i <-> lane i^16 (identity in-row mapping) ----
__device__ __forceinline__ int xswap_i(int x) {
    return __builtin_amdgcn_permlanex16(x, x, 0x76543210, 0xfedcba98, false, false);
}
__device__ __forceinline__ float xswap_f(float x) {
    return __builtin_bit_cast(float, xswap_i(__builtin_bit_cast(int, x)));
}
// pack two f32 -> packed 2xf16 in one dword
__device__ __forceinline__ int pack2h(float a, float b) {
    union { _Float16 h[2]; int i; } u;
    u.h[0] = (_Float16)a; u.h[1] = (_Float16)b;
    return u.i;
}
// CDNA5 async global->LDS 16-byte copy (ASYNCcnt-tracked, no VGPR round-trip)
__device__ __forceinline__ void async_copy_b128(unsigned int lds_off, const void* gptr) {
    asm volatile("global_load_async_to_lds_b128 %0, %1, off"
                 :: "v"(lds_off), "v"((unsigned long long)(uintptr_t)gptr)
                 : "memory");
}
__device__ __forceinline__ void wait_async0() {
    asm volatile("s_wait_asynccnt 0x0" ::: "memory");
}

// =================== pre-pass: K -> f16, V -> f16 transposed ===================
// one block per (bh, 64-key tile); converts once instead of once per q-block (16x)
__global__ __launch_bounds__(256)
void cvt_kv_kernel(const float* __restrict__ kk, const float* __restrict__ vv,
                   _Float16* __restrict__ kh, _Float16* __restrict__ vth)
{
    __shared__ __align__(32) _Float16 TV[HEAD_D][KT + LPAD];

    const int tid   = threadIdx.x;
    const int bh    = blockIdx.x >> 5;     // 0..31
    const int ktile = blockIdx.x & 31;     // 0..31
    const int key0  = ktile * KT;
    const int lkey  = tid & 63;
    const int ld0   = (tid >> 6) * 16;

    const float4* krow = (const float4*)(kk + ((size_t)bh * S_LEN + key0 + lkey) * HEAD_D + ld0);
    const float4* vrow = (const float4*)(vv + ((size_t)bh * S_LEN + key0 + lkey) * HEAD_D + ld0);
    v8h klo, khi;
    #pragma unroll
    for (int j = 0; j < 4; ++j) {
        float4 kf = krow[j], vf = vrow[j];
        _Float16 k0 = (_Float16)kf.x, k1 = (_Float16)kf.y,
                 k2 = (_Float16)kf.z, k3 = (_Float16)kf.w;
        if (j < 2) { klo[4*j] = k0; klo[4*j+1] = k1; klo[4*j+2] = k2; klo[4*j+3] = k3; }
        else       { khi[4*(j-2)] = k0; khi[4*(j-2)+1] = k1; khi[4*(j-2)+2] = k2; khi[4*(j-2)+3] = k3; }
        TV[ld0 + 4*j    ][lkey] = (_Float16)vf.x;
        TV[ld0 + 4*j + 1][lkey] = (_Float16)vf.y;
        TV[ld0 + 4*j + 2][lkey] = (_Float16)vf.z;
        TV[ld0 + 4*j + 3][lkey] = (_Float16)vf.w;
    }
    _Float16* ko = kh + ((size_t)bh * S_LEN + key0 + lkey) * HEAD_D + ld0;
    *(v8h*)ko       = klo;
    *(v8h*)(ko + 8) = khi;
    __syncthreads();
    // coalesced transposed store: vth[bh][d][key]
    _Float16* vo = vth + ((size_t)bh * HEAD_D + lkey) * S_LEN + key0 + ld0;
    *(v8h*)vo       = *(const v8h*)&TV[lkey][ld0];
    *(v8h*)(vo + 8) = *(const v8h*)&TV[lkey][ld0 + 8];
}

// =================== main flash-attention kernel ===================
__global__ __launch_bounds__(256)
void fa_fwd_kernel(const float* __restrict__ q, const _Float16* __restrict__ kh,
                   const _Float16* __restrict__ vth, const float* __restrict__ mask,
                   float* __restrict__ out)
{
    // Double-buffered LDS tiles, filled by async global->LDS copies
    __shared__ __align__(32) _Float16 Klds[2][KT][HEAD_D + LPAD];
    __shared__ __align__(32) _Float16 Vt[2][HEAD_D][KT + LPAD];

    const int tid  = threadIdx.x;
    const int wave = tid >> 5;
    const int lane = tid & 31;
    const int hlf  = lane >> 4;
    const int n    = lane & 15;      // qrow (B/C column) and A-row index
    const int kb   = hlf * 8;

    const int bh    = blockIdx.x >> 4;
    const int qblk  = blockIdx.x & 15;
    const int qbase = qblk * (16 * WAVES) + wave * 16;

    const float*    Qg = q   + (size_t)bh * S_LEN * HEAD_D;
    const _Float16* Kh = kh  + (size_t)bh * S_LEN * HEAD_D;
    const _Float16* Vh = vth + (size_t)bh * HEAD_D * S_LEN;
    float*          Og = out + (size_t)bh * S_LEN * HEAD_D;

    // ---- Q^T B-fragments (32 d x 16 qrows each): lane = qrow, halves = d ----
    v16h bq0, bq1;
    {
        const float* qr = Qg + (size_t)(qbase + n) * HEAD_D + hlf * 16;
        const float qscale = 0.125f; // 1/sqrt(64)
        #pragma unroll
        for (int j = 0; j < 16; ++j) {
            bq0[j] = (_Float16)(qr[j]      * qscale);
            bq1[j] = (_Float16)(qr[32 + j] * qscale);
        }
    }

    v8f acc[4] = {{}, {}, {}, {}};
    float mrow = -3.0e38f, lrow = 0.0f;
    const float* maskp = mask + (size_t)(qbase + n) * S_LEN;

    // staging partition: 256 threads x 2 b128 per tensor = 8 KB per tensor tile
    const int lkey = tid & 63;           // K row / Vt row
    const int ld0  = (tid >> 6) * 16;    // 16-half chunk

    auto issue_async = [&](int kblk, int buf) {
        const _Float16* ks = Kh + (size_t)(kblk + lkey) * HEAD_D + ld0;
        unsigned int kd = (unsigned int)(uintptr_t)&Klds[buf][lkey][ld0];
        async_copy_b128(kd,      ks);
        async_copy_b128(kd + 16, ks + 8);
        const _Float16* vs = Vh + (size_t)lkey * S_LEN + kblk + ld0;
        unsigned int vd = (unsigned int)(uintptr_t)&Vt[buf][lkey][ld0];
        async_copy_b128(vd,      vs);
        async_copy_b128(vd + 16, vs + 8);
    };

    issue_async(0, 0);   // prologue: tile 0 in flight

    for (int it = 0; it < S_LEN / KT; ++it) {
        const int kblk = it * KT;
        const int buf  = it & 1;

        wait_async0();       // this wave's copies into `buf` are done
        __syncthreads();     // everyone's copies are done

        if (kblk + KT < S_LEN) issue_async(kblk + KT, buf ^ 1);
        if (kblk + 2 * KT < S_LEN) {
            __builtin_prefetch(Kh + (size_t)(kblk + 2 * KT + lkey) * HEAD_D + ld0, 0, 0);
            __builtin_prefetch(Vh + (size_t)lkey * S_LEN + kblk + 2 * KT + ld0, 0, 0);
        }

        const _Float16 (*Kb)[HEAD_D + LPAD] = Klds[buf];
        const _Float16 (*Vb)[KT + LPAD]     = Vt[buf];

        // ---- S^T tiles: c[t] = K_tile(16 keys x 64 d) * Q^T  (8 WMMA) ----
        v8f c[4];
        #pragma unroll
        for (int t = 0; t < 4; ++t) {
            const _Float16* kr = &Kb[16 * t + n][0];
            v8h a0l = *(const v8h*)(kr + 2 * kb),      a0h = *(const v8h*)(kr + 8 + 2 * kb);
            v8h a1l = *(const v8h*)(kr + 32 + 2 * kb), a1h = *(const v8h*)(kr + 40 + 2 * kb);
            v16h a0, a1;
            #pragma unroll
            for (int j = 0; j < 8; ++j) { a0[j] = a0l[j]; a0[8+j] = a0h[j];
                                          a1[j] = a1l[j]; a1[8+j] = a1h[j]; }
            v8f z = {};
            z = __builtin_amdgcn_wmma_f32_16x16x32_f16(false, a0, false, bq0, (short)0, z, false, false);
            z = __builtin_amdgcn_wmma_f32_16x16x32_f16(false, a1, false, bq1, (short)0, z, false, false);
            c[t] = z;
        }

        // ---- mask + online softmax ----
        float sv[4][8];
        float lmax = -3.0e38f;
        #pragma unroll
        for (int t = 0; t < 4; ++t) {
            float4 m0 = *(const float4*)(maskp + kblk + 16 * t + kb);
            float4 m1 = *(const float4*)(maskp + kblk + 16 * t + kb + 4);
            sv[t][0] = c[t][0] + m0.x; sv[t][1] = c[t][1] + m0.y;
            sv[t][2] = c[t][2] + m0.z; sv[t][3] = c[t][3] + m0.w;
            sv[t][4] = c[t][4] + m1.x; sv[t][5] = c[t][5] + m1.y;
            sv[t][6] = c[t][6] + m1.z; sv[t][7] = c[t][7] + m1.w;
            #pragma unroll
            for (int r = 0; r < 8; ++r) lmax = fmaxf(lmax, sv[t][r]);
        }
        lmax = fmaxf(lmax, xswap_f(lmax));
        const float mnew = fmaxf(mrow, lmax);

        float p[4][8], rs = 0.0f;
        #pragma unroll
        for (int t = 0; t < 4; ++t)
            #pragma unroll
            for (int r = 0; r < 8; ++r) { p[t][r] = __expf(sv[t][r] - mnew); rs += p[t][r]; }
        rs += xswap_f(rs);
        const float alpha = __expf(mrow - mnew);
        lrow = lrow * alpha + rs;
        mrow = mnew;
        #pragma unroll
        for (int t = 0; t < 4; ++t)
            #pragma unroll
            for (int r = 0; r < 8; ++r) acc[t][r] *= alpha;

        // ---- build P^T B-fragments: pack f16 pairs, half-wave swap, select ----
        int pk[4][4], sw[4][4];
        #pragma unroll
        for (int t = 0; t < 4; ++t)
            #pragma unroll
            for (int jj = 0; jj < 4; ++jj) {
                pk[t][jj] = pack2h(p[t][2*jj], p[t][2*jj+1]);
                sw[t][jj] = xswap_i(pk[t][jj]);
            }
        const bool lowhalf = (hlf == 0);
        v16h bp[2];
        #pragma unroll
        for (int s = 0; s < 2; ++s) {
            v8i bi;
            #pragma unroll
            for (int jj = 0; jj < 4; ++jj) {
                bi[jj]     = lowhalf ? pk[2*s][jj] : sw[2*s + 1][jj];
                bi[4 + jj] = lowhalf ? sw[2*s][jj] : pk[2*s + 1][jj];
            }
            bp[s] = __builtin_bit_cast(v16h, bi);
        }

        // ---- O^T += V^T(16 d x 64 keys) * P^T  (8 WMMA) ----
        #pragma unroll
        for (int t = 0; t < 4; ++t) {
            const _Float16* vr = &Vb[16 * t + n][0];
            v8h a0l = *(const v8h*)(vr + 2 * kb),      a0h = *(const v8h*)(vr + 8 + 2 * kb);
            v8h a1l = *(const v8h*)(vr + 32 + 2 * kb), a1h = *(const v8h*)(vr + 40 + 2 * kb);
            v16h a0, a1;
            #pragma unroll
            for (int j = 0; j < 8; ++j) { a0[j] = a0l[j]; a0[8+j] = a0h[j];
                                          a1[j] = a1l[j]; a1[8+j] = a1h[j]; }
            acc[t] = __builtin_amdgcn_wmma_f32_16x16x32_f16(false, a0, false, bp[0], (short)0, acc[t], false, false);
            acc[t] = __builtin_amdgcn_wmma_f32_16x16x32_f16(false, a1, false, bp[1], (short)0, acc[t], false, false);
        }
    }

    // ---- normalize and write O ----
    {
        const float inv = 1.0f / lrow;
        float* orow = Og + (size_t)(qbase + n) * HEAD_D;
        #pragma unroll
        for (int t = 0; t < 4; ++t) {
            float4 f0, f1;
            f0.x = acc[t][0] * inv; f0.y = acc[t][1] * inv;
            f0.z = acc[t][2] * inv; f0.w = acc[t][3] * inv;
            f1.x = acc[t][4] * inv; f1.y = acc[t][5] * inv;
            f1.z = acc[t][6] * inv; f1.w = acc[t][7] * inv;
            *(float4*)(orow + 16 * t + kb)     = f0;
            *(float4*)(orow + 16 * t + kb + 4) = f1;
        }
    }
}

extern "C" void kernel_launch(void* const* d_in, const int* in_sizes, int n_in,
                              void* d_out, int out_size, void* d_ws, size_t ws_size,
                              hipStream_t stream) {
    (void)in_sizes; (void)n_in; (void)out_size; (void)ws_size;
    const float* q    = (const float*)d_in[0];
    const float* k    = (const float*)d_in[1];
    const float* v    = (const float*)d_in[2];
    const float* mask = (const float*)d_in[3];
    float* out = (float*)d_out;

    // workspace: f16 K [bh][key][d] and f16 V^T [bh][d][key]  (8.4 MB each)
    _Float16* kh  = (_Float16*)d_ws;
    _Float16* vth = kh + (size_t)BH * S_LEN * HEAD_D;

    cvt_kv_kernel<<<dim3(BH * (S_LEN / KT)), dim3(256), 0, stream>>>(k, v, kh, vth);
    fa_fwd_kernel<<<dim3(512), dim3(256), 0, stream>>>(q, kh, vth, mask, out);
}
